// SimGRewGAT_29772713296408
// MI455X (gfx1250) — compile-verified, matrix operand order checked by reading, not run on hardware
//
#include <hip/hip_runtime.h>
#include <hip/hip_bf16.h>
#include <math.h>

// ---- problem constants (match reference) ----
#define NN      4096     // nodes
#define F_IN    512
#define HID     128
#define HEADS   2
#define HC      256      // HEADS*HID
#define N_CLS   40
#define NEG_SLOPE 0.2f
#define LN_EPS  1e-5f

typedef __attribute__((ext_vector_type(16))) __bf16 v16bf;
typedef __attribute__((ext_vector_type(8)))  __bf16 v8bf;
typedef __attribute__((ext_vector_type(8)))  float  v8f;

// ---------------------------------------------------------------------------
// Small utility kernels
// ---------------------------------------------------------------------------
__global__ void k_zero(float* p, int n) {
    int i = blockIdx.x * blockDim.x + threadIdx.x;
    if (i < n) p[i] = 0.f;
}

// fp32 -> bf16, flat copy (row-major preserved)
__global__ void k_f2bf(const float* __restrict__ in, __bf16* __restrict__ out, int n) {
    int i = blockIdx.x * blockDim.x + threadIdx.x;
    if (i < n) out[i] = (__bf16)in[i];
}

// fp32 [R x C] row-major -> bf16 [C x R] (transpose), so that K-consecutive
// elements of a WMMA B-fragment are contiguous in memory.
__global__ void k_f2bf_tr(const float* __restrict__ in, __bf16* __restrict__ out,
                          int R, int C) {
    int i = blockIdx.x * blockDim.x + threadIdx.x;
    if (i >= R * C) return;
    int r = i / C, c = i % C;
    out[(size_t)c * R + r] = (__bf16)in[i];
}

// per-row adjacency sums: rs[i]=rsqrt(rowsum+1); nd[i]=nnz+2; totals via atomics
__global__ void k_degree(const float* __restrict__ adj, float* rs, float* nd, float* scal) {
    __shared__ float sS[256], sC[256];
    int row = blockIdx.x, tid = threadIdx.x;
    const float* r = adj + (size_t)row * NN;
    float s = 0.f, c = 0.f;
    for (int j0 = tid * 4; j0 < NN; j0 += 1024) {
        float4 a = *(const float4*)(r + j0);
        s += a.x + a.y + a.z + a.w;
        c += ((a.x != 0.f) ? 1.f : 0.f) + ((a.y != 0.f) ? 1.f : 0.f) +
             ((a.z != 0.f) ? 1.f : 0.f) + ((a.w != 0.f) ? 1.f : 0.f);
    }
    sS[tid] = s; sC[tid] = c; __syncthreads();
    for (int off = 128; off > 0; off >>= 1) {
        if (tid < off) { sS[tid] += sS[tid + off]; sC[tid] += sC[tid + off]; }
        __syncthreads();
    }
    if (tid == 0) {
        rs[row] = rsqrtf(sS[0] + 1.f);          // D^{-1/2} of (A+I)
        nd[row] = sC[0] + 2.f;                  // mask rowsum (+self) + 1
        atomicAdd(&scal[1], sS[0]);             // adj.sum()
        atomicAdd(&scal[2], sC[0]);             // count_nonzero(adj)
    }
}

// norm_adj[i,j] = (adj+I)[i,j]*rs[i]*rs[j]; accumulate scale=sum(norm_adj)
__global__ void k_normadj(const float* __restrict__ adj, const float* __restrict__ rs,
                          float* __restrict__ na, float* scal) {
    __shared__ float sS[256];
    int row = blockIdx.x, tid = threadIdx.x;
    const float* r = adj + (size_t)row * NN;
    float ri = rs[row], s = 0.f;
    for (int j0 = tid * 4; j0 < NN; j0 += 1024) {
        float4 a = *(const float4*)(r + j0);
        float4 q = *(const float4*)(rs + j0);
        float v0 = (a.x + ((j0 + 0 == row) ? 1.f : 0.f)) * ri * q.x;
        float v1 = (a.y + ((j0 + 1 == row) ? 1.f : 0.f)) * ri * q.y;
        float v2 = (a.z + ((j0 + 2 == row) ? 1.f : 0.f)) * ri * q.z;
        float v3 = (a.w + ((j0 + 3 == row) ? 1.f : 0.f)) * ri * q.w;
        na[(size_t)row * NN + j0 + 0] = v0;
        na[(size_t)row * NN + j0 + 1] = v1;
        na[(size_t)row * NN + j0 + 2] = v2;
        na[(size_t)row * NN + j0 + 3] = v3;
        s += v0 + v1 + v2 + v3;
    }
    sS[tid] = s; __syncthreads();
    for (int off = 128; off > 0; off >>= 1) {
        if (tid < off) sS[tid] += sS[tid + off];
        __syncthreads();
    }
    if (tid == 0) atomicAdd(&scal[0], sS[0]);
}

// ---------------------------------------------------------------------------
// WMMA GEMM on pre-converted bf16: C[MxNc](f32) = A[MxK](bf16,row-major)
//                                             @ Bt[NcxK](bf16, B transposed).
// Per CDNA5 16-bit fragment layout, a lane's 16 elements are two contiguous
// 8-element bf16 runs -> two b128 loads per matrix per K-step. Inner loop is
// 4x global_load_b128 + 1x v_wmma per 32-K step.
// ---------------------------------------------------------------------------
__global__ void k_gemm_wmma(const __bf16* __restrict__ A, const __bf16* __restrict__ Bt,
                            float* __restrict__ C, int K, int Nc) {
    int lane = threadIdx.x;
    int j    = lane & 15;
    int half = lane >> 4;
    int m0 = blockIdx.x * 16;
    int n0 = blockIdx.y * 16;
    int am = m0 + j;     // A row handled by this lane
    int bn = n0 + j;     // B column handled by this lane
    const __bf16* Arow = A  + (size_t)am * K;
    const __bf16* Bcol = Bt + (size_t)bn * K;
    v8f acc = {};
    for (int kk = 0; kk < K; kk += 32) {
        __builtin_prefetch(Arow + kk + 32, 0, 3);
        v8bf alo = *(const v8bf*)(Arow + kk + half * 8);        // K run [kk+h8, +8)
        v8bf ahi = *(const v8bf*)(Arow + kk + 16 + half * 8);   // K run [kk+16+h8, +8)
        v8bf blo = *(const v8bf*)(Bcol + kk + half * 8);
        v8bf bhi = *(const v8bf*)(Bcol + kk + 16 + half * 8);
        v16bf a = __builtin_shufflevector(alo, ahi, 0,1,2,3,4,5,6,7,8,9,10,11,12,13,14,15);
        v16bf b = __builtin_shufflevector(blo, bhi, 0,1,2,3,4,5,6,7,8,9,10,11,12,13,14,15);
        acc = __builtin_amdgcn_wmma_f32_16x16x32_bf16(false, a, false, b,
                                                      (short)0, acc, false, false);
    }
#pragma unroll
    for (int v = 0; v < 8; ++v) {
        int m = m0 + v + 8 * half;
        C[(size_t)m * Nc + bn] = acc[v];
    }
}

// attention logits: es[n,h]=xp[n,h,:].a_src[h,:], ed likewise
__global__ void k_logits(const float* __restrict__ xp, const float* __restrict__ a_src,
                         const float* __restrict__ a_dst, float* es, float* ed, int C) {
    int n = blockIdx.x * blockDim.x + threadIdx.x;
    if (n >= NN) return;
    for (int h = 0; h < HEADS; ++h) {
        const float* row = xp + (size_t)n * (HEADS * C) + h * C;
        float s = 0.f, d = 0.f;
        for (int c = 0; c < C; ++c) {
            float v = row[c];
            s += v * a_src[h * C + c];
            d += v * a_dst[h * C + c];
        }
        es[n * HEADS + h] = s;
        ed[n * HEADS + h] = d;
    }
}

// masked softmax-over-sources stats per target t (both heads): m[t,h], den[t,h]
__global__ void k_softmax(const float* __restrict__ adj, const float* __restrict__ es,
                          const float* __restrict__ ed, float* mOut, float* dOut) {
    __shared__ float r0[256], r1[256];
    int t = blockIdx.x, tid = threadIdx.x;
    const float* row = adj + (size_t)t * NN;
    float ed0 = ed[t * HEADS + 0], ed1 = ed[t * HEADS + 1];
    float m0 = -__builtin_inff(), m1 = -__builtin_inff();
    for (int j0 = tid * 4; j0 < NN; j0 += 1024) {
        float4 a = *(const float4*)(row + j0);
#pragma unroll
        for (int u = 0; u < 4; ++u) {
            float av = (u == 0) ? a.x : (u == 1) ? a.y : (u == 2) ? a.z : a.w;
            int s = j0 + u;
            if (av != 0.f || s == t) {
                float e0 = es[s * HEADS] + ed0;     e0 = e0 > 0.f ? e0 : NEG_SLOPE * e0;
                float e1 = es[s * HEADS + 1] + ed1; e1 = e1 > 0.f ? e1 : NEG_SLOPE * e1;
                m0 = fmaxf(m0, e0); m1 = fmaxf(m1, e1);
            }
        }
    }
    r0[tid] = m0; r1[tid] = m1; __syncthreads();
    for (int off = 128; off > 0; off >>= 1) {
        if (tid < off) { r0[tid] = fmaxf(r0[tid], r0[tid + off]);
                         r1[tid] = fmaxf(r1[tid], r1[tid + off]); }
        __syncthreads();
    }
    m0 = r0[0]; m1 = r1[0]; __syncthreads();
    float d0 = 0.f, d1 = 0.f;
    for (int j0 = tid * 4; j0 < NN; j0 += 1024) {
        float4 a = *(const float4*)(row + j0);
#pragma unroll
        for (int u = 0; u < 4; ++u) {
            float av = (u == 0) ? a.x : (u == 1) ? a.y : (u == 2) ? a.z : a.w;
            int s = j0 + u;
            if (av != 0.f || s == t) {
                float e0 = es[s * HEADS] + ed0;     e0 = e0 > 0.f ? e0 : NEG_SLOPE * e0;
                float e1 = es[s * HEADS + 1] + ed1; e1 = e1 > 0.f ? e1 : NEG_SLOPE * e1;
                d0 += __expf(e0 - m0); d1 += __expf(e1 - m1);
            }
        }
    }
    r0[tid] = d0; r1[tid] = d1; __syncthreads();
    for (int off = 128; off > 0; off >>= 1) {
        if (tid < off) { r0[tid] += r0[tid + off]; r1[tid] += r1[tid + off]; }
        __syncthreads();
    }
    if (tid == 0) {
        mOut[t * HEADS] = m0; mOut[t * HEADS + 1] = m1;
        dOut[t * HEADS] = r0[0]; dOut[t * HEADS + 1] = r1[0];
    }
}

// Precompute per-node alpha factors so the WMMA K-loops need no transcendentals:
//   Ep[h][n] = { exp(es), exp(0.2*es) }                  (source side, both heads)
//   Fp[n]    = { ed, exp(ed-m)/den, exp(0.2*ed-m)/den }  (target side, head 0)
// alpha(s->t) = mask * ( es_s+ed_t>0 ? E1_s*c1_t : E2_s*c2_t ), and the sign
// test es_s > -ed_t  <=>  E1_s > exp(-ed_t) by monotonicity of exp.
__global__ void k_prep(const float* __restrict__ es, const float* __restrict__ ed,
                       const float* __restrict__ mm, const float* __restrict__ den,
                       float2* __restrict__ Ep, float4* __restrict__ Fp) {
    int n = blockIdx.x * blockDim.x + threadIdx.x;
    if (n >= NN) return;
#pragma unroll
    for (int h = 0; h < HEADS; ++h) {
        float e = es[n * HEADS + h];
        Ep[(size_t)h * NN + n] = make_float2(__expf(e), __expf(NEG_SLOPE * e));
    }
    float edv = ed[n * HEADS], mv = mm[n * HEADS], inv = 1.f / den[n * HEADS];
    Fp[n] = make_float4(edv, __expf(edv - mv) * inv,
                        __expf(NEG_SLOPE * edv - mv) * inv, 0.f);
}

// ---------------------------------------------------------------------------
// WMMA aggregation: out[m, col] = sum_k alpha(m,k) * Bt[bcol, k] (+bias[col])
//   transposed==0 (layer agg): m=target t, k=source s
//   transposed==1 (We @ gf)  : m=source s, k=target t  (head 0)
// A side: alpha rebuilt per element pair from Ep/Fp (b64 adj + b128 table load,
// select/mul, packed cvt) -- zero exps in the K loop. B side: pre-transposed
// bf16 panel -> two b128 loads per K-step. adj stays L2-resident (67MB<192MB).
// ---------------------------------------------------------------------------
__global__ void k_agg_wmma(const float* __restrict__ adj,
                           const float* __restrict__ es, const float* __restrict__ ed,
                           const float* __restrict__ mm, const float* __restrict__ den,
                           const float2* __restrict__ Ep, const float4* __restrict__ Fp,
                           const __bf16* __restrict__ Bt,
                           float* __restrict__ out, int outStride,
                           const float* __restrict__ bias,
                           int headDiv, int validC, int transposed) {
    int lane = threadIdx.x;
    int j    = lane & 15;
    int half = lane >> 4;
    int m0  = blockIdx.x * 16;
    int n0  = blockIdx.y * 16;
    int am  = m0 + j;                    // output row handled by this lane (A side)
    int col = n0 + j;                    // output column handled by this lane (B side)
    int h   = n0 / headDiv;              // head is uniform within a 16-col tile
    int cB  = col % headDiv;
    int bcol = h * validC + cB;
    bool bvalid = (cB < validC);
    const float*  adjRow = adj + (size_t)am * NN;
    const float2* EpH    = Ep + (size_t)h * NN;
    const __bf16* Bcol   = Bt + (size_t)(bvalid ? bcol : 0) * NN;

    float c1 = 0.f, c2 = 0.f, T = 0.f, r1 = 0.f, r2 = 0.f, thr = 0.f;
    if (!transposed) {
        float edv = ed[am * HEADS + h], mv = mm[am * HEADS + h];
        float inv = 1.f / den[am * HEADS + h];
        c1 = __expf(edv - mv) * inv;
        c2 = __expf(NEG_SLOPE * edv - mv) * inv;
        T  = __expf(-edv);
    } else {
        float esv = es[am * HEADS + h];
        r1 = __expf(esv); r2 = __expf(NEG_SLOPE * esv); thr = -esv;
    }

    v8f acc = {};
    for (int kk = 0; kk < NN; kk += 32) {
        __builtin_prefetch(adjRow + kk + 32, 0, 3);
        // B fragment: two contiguous bf16 runs (zeroed for padding columns)
        v8bf blo = {}, bhi = {};
        if (bvalid) {
            blo = *(const v8bf*)(Bcol + kk + half * 8);
            bhi = *(const v8bf*)(Bcol + kk + 16 + half * 8);
        }
        v16bf b = __builtin_shufflevector(blo, bhi, 0,1,2,3,4,5,6,7,8,9,10,11,12,13,14,15);
        // A fragment: alpha reconstructed on the fly
        v16bf a;
#pragma unroll
        for (int v = 0; v < 8; ++v) {
            int k0 = kk + ((v & 4) << 2) + half * 8 + ((v & 3) << 1);
            float2 a2 = *(const float2*)(adjRow + k0);   // k0 even -> 8B aligned
            bool mk0 = (a2.x != 0.f) || (k0 == am);
            bool mk1 = (a2.y != 0.f) || (k0 + 1 == am);
            float al0, al1;
            if (!transposed) {
                float4 ee = *(const float4*)(EpH + k0);  // E1,E2 for k0 and k0+1
                al0 = mk0 ? (ee.x > T ? ee.x * c1 : ee.y * c2) : 0.f;
                al1 = mk1 ? (ee.z > T ? ee.z * c1 : ee.w * c2) : 0.f;
            } else {
                float4 f0 = Fp[k0], f1 = Fp[k0 + 1];
                al0 = mk0 ? (f0.x > thr ? r1 * f0.y : r2 * f0.z) : 0.f;
                al1 = mk1 ? (f1.x > thr ? r1 * f1.y : r2 * f1.z) : 0.f;
            }
            a[2 * v]     = (__bf16)al0;
            a[2 * v + 1] = (__bf16)al1;
        }
        acc = __builtin_amdgcn_wmma_f32_16x16x32_bf16(false, a, false, b,
                                                      (short)0, acc, false, false);
    }
    float bv = bias ? bias[col] : 0.f;
#pragma unroll
    for (int v = 0; v < 8; ++v) {
        int m = m0 + v + 8 * half;
        out[(size_t)m * outStride + col] = acc[v] + bv;
    }
}

// ELU + LayerNorm in place (one 256-thread block per row, HC==256)
__global__ void k_eluln(float* __restrict__ hbuf, const float* __restrict__ g,
                        const float* __restrict__ be) {
    __shared__ float red[256];
    int row = blockIdx.x, tid = threadIdx.x;
    float v = hbuf[(size_t)row * HC + tid];
    v = v > 0.f ? v : expm1f(v);                  // ELU(alpha=1)
    red[tid] = v; __syncthreads();
    for (int off = 128; off > 0; off >>= 1) {
        if (tid < off) red[tid] += red[tid + off];
        __syncthreads();
    }
    float mean = red[0] * (1.f / HC); __syncthreads();
    float d = v - mean;
    red[tid] = d * d; __syncthreads();
    for (int off = 128; off > 0; off >>= 1) {
        if (tid < off) red[tid] += red[tid + off];
        __syncthreads();
    }
    float var = red[0] * (1.f / HC);
    hbuf[(size_t)row * HC + tid] = d * rsqrtf(var + LN_EPS) * g[tid] + be[tid];
}

// layer2: mean over heads + bias -> final h
__global__ void k_headmean(const float* __restrict__ agg, const float* __restrict__ b2,
                           float* __restrict__ hOut) {
    int i = blockIdx.x * blockDim.x + threadIdx.x;
    if (i >= NN * N_CLS) return;
    int t = i / N_CLS, c = i % N_CLS;
    hOut[i] = 0.5f * (agg[(size_t)t * 128 + c] + agg[(size_t)t * 128 + 64 + c]) + b2[c];
}

// log_softmax over classes (64-thread block per row)
__global__ void k_logsoftmax(const float* __restrict__ h, float* __restrict__ logp) {
    __shared__ float red[64];
    int t = blockIdx.x, tid = threadIdx.x;
    float v = (tid < N_CLS) ? h[(size_t)t * N_CLS + tid] : -__builtin_inff();
    red[tid] = v; __syncthreads();
    for (int off = 32; off > 0; off >>= 1) {
        if (tid < off) red[tid] = fmaxf(red[tid], red[tid + off]);
        __syncthreads();
    }
    float mx = red[0]; __syncthreads();
    float ex = (tid < N_CLS) ? __expf(v - mx) : 0.f;
    red[tid] = ex; __syncthreads();
    for (int off = 32; off > 0; off >>= 1) {
        if (tid < off) red[tid] += red[tid + off];
        __syncthreads();
    }
    float lse = __logf(red[0]) + mx;
    if (tid < N_CLS) logp[(size_t)t * N_CLS + tid] = v - lse;
}

// gf = relu(h) * rsqrt(nd)[:,None] (row stride 64, zero padded); sq = |gf_row|^2
__global__ void k_gf(const float* __restrict__ h, const float* __restrict__ nd,
                     float* __restrict__ gf, float* __restrict__ sq) {
    __shared__ float red[64];
    int t = blockIdx.x, tid = threadIdx.x;
    float r = rsqrtf(nd[t]);
    float f = 0.f;
    if (tid < N_CLS) {
        float v = h[(size_t)t * N_CLS + tid];
        f = (v > 0.f ? v : 0.f) * r;
    }
    gf[(size_t)t * 64 + tid] = f;
    red[tid] = f * f; __syncthreads();
    for (int off = 32; off > 0; off >>= 1) {
        if (tid < off) red[tid] += red[tid + off];
        __syncthreads();
    }
    if (tid == 0) sq[t] = red[0];
}

// rw[i] = sum_t alpha2(i->t, head0); cw[i] = sum_s alpha2(s->i, head0)
// Uses Ep/Fp tables: no exp in the scan loop.
__global__ void k_rwcw(const float* __restrict__ adj, const float* __restrict__ es,
                       const float* __restrict__ ed, const float* __restrict__ mm,
                       const float* __restrict__ den,
                       const float2* __restrict__ Ep, const float4* __restrict__ Fp,
                       float* rw, float* cw) {
    __shared__ float rA[256], rB[256];
    int i = blockIdx.x, tid = threadIdx.x;
    float esi = es[i * HEADS], edv = ed[i * HEADS];
    float mi = mm[i * HEADS], invi = 1.f / den[i * HEADS];
    float r1 = __expf(esi), r2 = __expf(NEG_SLOPE * esi), thr = -esi;      // rw side
    float c1 = __expf(edv - mi) * invi;
    float c2 = __expf(NEG_SLOPE * edv - mi) * invi;
    float T  = __expf(-edv);                                               // cw side
    const float* row = adj + (size_t)i * NN;
    float sr = 0.f, sc = 0.f;
    for (int j0 = tid * 4; j0 < NN; j0 += 1024) {
        float4 a = *(const float4*)(row + j0);
#pragma unroll
        for (int u = 0; u < 4; ++u) {
            float av = (u == 0) ? a.x : (u == 1) ? a.y : (u == 2) ? a.z : a.w;
            int j2 = j0 + u;
            if (av != 0.f || j2 == i) {
                float4 f = Fp[j2];                       // alpha(s=i, t=j2)
                sr += (f.x > thr) ? r1 * f.y : r2 * f.z;
                float2 e = Ep[j2];                       // head0: alpha(s=j2, t=i)
                sc += (e.x > T) ? e.x * c1 : e.y * c2;
            }
        }
    }
    rA[tid] = sr; rB[tid] = sc; __syncthreads();
    for (int off = 128; off > 0; off >>= 1) {
        if (tid < off) { rA[tid] += rA[tid + off]; rB[tid] += rB[tid + off]; }
        __syncthreads();
    }
    if (tid == 0) { rw[i] = rA[0]; cw[i] = rB[0]; }
}

// scal[3] += rw.sq ; scal[4] += cw.sq
__global__ void k_rcdot(const float* rw, const float* cw, const float* sq, float* scal) {
    __shared__ float rA[256], rB[256];
    int tid = threadIdx.x;
    float a = 0.f, b = 0.f;
    for (int i = blockIdx.x * 256 + tid; i < NN; i += gridDim.x * 256) {
        a += rw[i] * sq[i];
        b += cw[i] * sq[i];
    }
    rA[tid] = a; rB[tid] = b; __syncthreads();
    for (int off = 128; off > 0; off >>= 1) {
        if (tid < off) { rA[tid] += rA[tid + off]; rB[tid] += rB[tid + off]; }
        __syncthreads();
    }
    if (tid == 0) { atomicAdd(&scal[3], rA[0]); atomicAdd(&scal[4], rB[0]); }
}

// scal[5] += sum(y * gf)  (valid classes only)
__global__ void k_crossdot(const float* __restrict__ y, const float* __restrict__ gf,
                           float* scal) {
    __shared__ float red[256];
    int tid = threadIdx.x;
    float s = 0.f;
    for (int i = blockIdx.x * 256 + tid; i < NN * N_CLS; i += gridDim.x * 256) {
        int t = i / N_CLS, c = i % N_CLS;
        s += y[(size_t)t * 64 + c] * gf[(size_t)t * 64 + c];
    }
    red[tid] = s; __syncthreads();
    for (int off = 128; off > 0; off >>= 1) {
        if (tid < off) red[tid] += red[tid + off];
        __syncthreads();
    }
    if (tid == 0) atomicAdd(&scal[5], red[0]);
}

__global__ void k_final(const float* scal, const float* prob, float* outScalars) {
    float scale = scal[0], adjsum = scal[1], nnz = scal[2];
    float rwsq = scal[3], cwsq = scal[4], cross = scal[5];
    float de = 0.5f * (rwsq + cwsq - 2.f * cross);
    outScalars[0] = (scale != 0.f) ? de / scale : 0.f;   // dir_energy
    outScalars[1] = prob[0];                             // prob
    outScalars[2] = nnz / adjsum;                        // edge_ratio
}

// ---------------------------------------------------------------------------
extern "C" void kernel_launch(void* const* d_in, const int* in_sizes, int n_in,
                              void* d_out, int out_size, void* d_ws, size_t ws_size,
                              hipStream_t stream) {
    (void)in_sizes; (void)n_in; (void)out_size; (void)ws_size;
    const float* x   = (const float*)d_in[0];
    const float* adj = (const float*)d_in[1];
    const float* W0  = (const float*)d_in[2];
    const float* as0 = (const float*)d_in[3];
    const float* ad0 = (const float*)d_in[4];
    const float* b0  = (const float*)d_in[5];
    const float* g0  = (const float*)d_in[6];
    const float* be0 = (const float*)d_in[7];
    const float* W1  = (const float*)d_in[8];
    const float* as1 = (const float*)d_in[9];
    const float* ad1 = (const float*)d_in[10];
    const float* b1  = (const float*)d_in[11];
    const float* g1  = (const float*)d_in[12];
    const float* be1 = (const float*)d_in[13];
    const float* W2  = (const float*)d_in[14];
    const float* as2 = (const float*)d_in[15];
    const float* ad2 = (const float*)d_in[16];
    const float* b2  = (const float*)d_in[17];
    const float* prob = (const float*)d_in[18];

    float* ws = (float*)d_ws;
    float*  rs   = ws + 0;                 // [N]
    float*  nd   = ws + 4096;              // [N]
    float*  es   = ws + 8192;              // [N,2]
    float*  ed   = ws + 16384;             // [N,2]
    float*  mrow = ws + 24576;             // [N,2]
    float*  den  = ws + 32768;             // [N,2]
    float*  scal = ws + 40960;             // [16]
    float*  sq   = ws + 49152;             // [N]
    float*  rw   = ws + 53248;             // [N]
    float*  cw   = ws + 57344;             // [N]
    float2* Ep   = (float2*)(ws + 61440);  // [HEADS][N]
    float4* Fp   = (float4*)(ws + 77824);  // [N]
    float*  xp   = ws + 98304;             // [N,256] fp32 (layer2: [N,80])
    float*  hA   = xp + (size_t)NN * HC;   // [N,256]
    float*  hB   = hA + (size_t)NN * HC;   // [N,256]
    float*  gf   = hB + (size_t)NN * HC;   // [N,64] padded
    float*  yb   = gf + (size_t)NN * 64;   // [N,64] padded
    // bf16 staging: activations (row-major), weights^T, B-panel^T
    __bf16* Abf  = (__bf16*)(ws + 3768320); // up to [N,512] bf16
    __bf16* Wbt  = (__bf16*)(ws + 4816896); // up to [256,512] bf16 (W^T)
    __bf16* BpT  = (__bf16*)(ws + 4882432); // up to [256,N] bf16 (xp^T / gf^T)

    float* out_h    = (float*)d_out;           // [N,40]
    float* out_logp = out_h + NN * N_CLS;      // [N,40]
    float* out_scal = out_logp + NN * N_CLS;   // [3]: de, prob, edge_ratio
    float* out_na   = out_scal + 3;            // [N,N]

    // ---- graph prep ----
    k_zero<<<1, 32, 0, stream>>>(scal, 16);
    k_degree<<<NN, 256, 0, stream>>>(adj, rs, nd, scal);
    k_normadj<<<NN, 256, 0, stream>>>(adj, rs, out_na, scal);

    // ---- layer 0: x[N,512] -> hA[N,256] ----
    k_f2bf<<<(NN * F_IN + 255) / 256, 256, 0, stream>>>(x, Abf, NN * F_IN);
    k_f2bf_tr<<<(F_IN * HC + 255) / 256, 256, 0, stream>>>(W0, Wbt, F_IN, HC);
    k_gemm_wmma<<<dim3(NN / 16, HC / 16), 32, 0, stream>>>(Abf, Wbt, xp, F_IN, HC);
    k_logits<<<NN / 256, 256, 0, stream>>>(xp, as0, ad0, es, ed, HID);
    k_softmax<<<NN, 256, 0, stream>>>(adj, es, ed, mrow, den);
    k_prep<<<NN / 256, 256, 0, stream>>>(es, ed, mrow, den, Ep, Fp);
    k_f2bf_tr<<<(NN * HC + 255) / 256, 256, 0, stream>>>(xp, BpT, NN, HC);
    k_agg_wmma<<<dim3(NN / 16, HC / 16), 32, 0, stream>>>(
        adj, es, ed, mrow, den, Ep, Fp, BpT, hA, HC, b0, HID, HID, 0);
    k_eluln<<<NN, HC, 0, stream>>>(hA, g0, be0);

    // ---- layer 1: hA -> hB ----
    k_f2bf<<<(NN * HC + 255) / 256, 256, 0, stream>>>(hA, Abf, NN * HC);
    k_f2bf_tr<<<(HC * HC + 255) / 256, 256, 0, stream>>>(W1, Wbt, HC, HC);
    k_gemm_wmma<<<dim3(NN / 16, HC / 16), 32, 0, stream>>>(Abf, Wbt, xp, HC, HC);
    k_logits<<<NN / 256, 256, 0, stream>>>(xp, as1, ad1, es, ed, HID);
    k_softmax<<<NN, 256, 0, stream>>>(adj, es, ed, mrow, den);
    k_prep<<<NN / 256, 256, 0, stream>>>(es, ed, mrow, den, Ep, Fp);
    k_f2bf_tr<<<(NN * HC + 255) / 256, 256, 0, stream>>>(xp, BpT, NN, HC);
    k_agg_wmma<<<dim3(NN / 16, HC / 16), 32, 0, stream>>>(
        adj, es, ed, mrow, den, Ep, Fp, BpT, hB, HC, b1, HID, HID, 0);
    k_eluln<<<NN, HC, 0, stream>>>(hB, g1, be1);

    // ---- layer 2: hB -> h (mean over heads) ----
    k_f2bf<<<(NN * HC + 255) / 256, 256, 0, stream>>>(hB, Abf, NN * HC);
    k_f2bf_tr<<<(HC * HEADS * N_CLS + 255) / 256, 256, 0, stream>>>(W2, Wbt, HC, HEADS * N_CLS);
    k_gemm_wmma<<<dim3(NN / 16, 80 / 16), 32, 0, stream>>>(Abf, Wbt, xp, HC, HEADS * N_CLS);
    k_logits<<<NN / 256, 256, 0, stream>>>(xp, as2, ad2, es, ed, N_CLS);
    k_softmax<<<NN, 256, 0, stream>>>(adj, es, ed, mrow, den);
    k_prep<<<NN / 256, 256, 0, stream>>>(es, ed, mrow, den, Ep, Fp);
    k_f2bf_tr<<<(NN * HEADS * N_CLS + 255) / 256, 256, 0, stream>>>(xp, BpT, NN, HEADS * N_CLS);
    // per-head agg into hA with padded layout: cols [h*64 + c], c < 40
    k_agg_wmma<<<dim3(NN / 16, 8), 32, 0, stream>>>(
        adj, es, ed, mrow, den, Ep, Fp, BpT, hA, 128, nullptr, 64, N_CLS, 0);
    k_headmean<<<(NN * N_CLS + 255) / 256, 256, 0, stream>>>(hA, b2, out_h);

    // ---- outputs / dirichlet energy ----
    k_logsoftmax<<<NN, 64, 0, stream>>>(out_h, out_logp);
    k_gf<<<NN, 64, 0, stream>>>(out_h, nd, gf, sq);
    k_rwcw<<<NN, 256, 0, stream>>>(adj, es, ed, mrow, den, Ep, Fp, rw, cw);
    k_rcdot<<<16, 256, 0, stream>>>(rw, cw, sq, scal);
    // cross = sum((We @ gf) * gf); y = We @ gf via transposed-alpha WMMA agg
    k_f2bf_tr<<<(NN * 64 + 255) / 256, 256, 0, stream>>>(gf, BpT, NN, 64);
    k_agg_wmma<<<dim3(NN / 16, 3), 32, 0, stream>>>(
        adj, es, ed, mrow, den, Ep, Fp, BpT, yb, 64, nullptr, 64, N_CLS, 1);
    k_crossdot<<<160, 256, 0, stream>>>(yb, gf, scal);
    k_final<<<1, 1, 0, stream>>>(scal, prob, out_scal);
}
